// MCLoss_45449343926802
// MI455X (gfx1250) — compile-verified, matrix operand order checked by reading, not run on hardware
//
#include <hip/hip_runtime.h>

// Problem sizes (fixed by the reference).
#define B_ROWS 4096
#define D_K    2048
#define C_COLS 50000

// Packed-path tiling: 8 waves/block; wave owns 32 M-rows x 80 N-cols
// (2 A-fragments x 5 B-fragments -> 10 WMMAs per k-step, 1 ds_load_b128/WMMA).
#define MTP 256                      // block M-tile (8 waves * 32 rows)
#define MTF 128                      // fallback block M-tile (8 waves * 16 rows)
#define NT 80
#define KC 128                       // K-chunk staged in LDS
#define LDSTRIDE (KC + 8)            // ushort elems per LDS row (pad: banks advance 4/row)
#define NITER (D_K / KC)             // 16
#define CHUNKS_PER_THREAD ((NT * (KC / 8)) / 256)   // 5

typedef __attribute__((ext_vector_type(16))) __bf16 v16bf;
typedef __attribute__((ext_vector_type(8)))  float  v8f;
typedef __attribute__((ext_vector_type(4)))  float  f32x4;
typedef __attribute__((ext_vector_type(4)))  unsigned int u32x4;

union Frag16 {                       // 32 bytes: one WMMA 16-bit A/B fragment
    u32x4          q[2];
    unsigned short h[16];
    v16bf          v;
};

// ---- CDNA5 async global->LDS path (ASYNCcnt), guarded so compile never breaks ----
#if __has_builtin(__builtin_amdgcn_global_load_async_to_lds_b128)
#define HAVE_ASYNC_LDS 1
#else
#define HAVE_ASYNC_LDS 0
#endif

typedef int v4i __attribute__((vector_size(4 * sizeof(int))));
typedef __attribute__((address_space(1))) v4i v4i_global;   // AS1: global ("__device__")
typedef __attribute__((address_space(3))) v4i v4i_lds;      // AS3: LDS ("__shared__")

__device__ __forceinline__ v4i_global* as_global_v4i(const void* p) {
    return (v4i_global*)(unsigned long long)p;
}
__device__ __forceinline__ v4i_lds* as_lds_v4i(void* p) {
    // generic LDS pointer: low 32 bits are the LDS byte offset (ISA aperture rule)
    return (v4i_lds*)(unsigned)(unsigned long long)p;
}
__device__ __forceinline__ void wait_asynccnt0() {
#if __has_builtin(__builtin_amdgcn_s_wait_asynccnt)
    __builtin_amdgcn_s_wait_asynccnt(0);
#else
    asm volatile("s_wait_asynccnt 0x0" ::: "memory");
#endif
}

__device__ __forceinline__ unsigned short f2bf(float f) {   // RNE fp32 -> bf16
    unsigned int u = __float_as_uint(f);
    u += 0x7FFFu + ((u >> 16) & 1u);
    return (unsigned short)(u >> 16);
}

// ---------------------------------------------------------------------------
// Pack fp32 -> bf16 into workspace (inputs: 16.8 MB, memory: 204.8 MB ~ L2)
// ---------------------------------------------------------------------------
__global__ void pack_bf16_kernel(const float* __restrict__ src,
                                 unsigned short* __restrict__ dst,
                                 long long n8) {
    long long i = (long long)blockIdx.x * blockDim.x + threadIdx.x;
    if (i >= n8) return;
    const f32x4* s = (const f32x4*)src + i * 2;
    f32x4 f0 = s[0];
    f32x4 f1 = s[1];
    union { unsigned short h[8]; u32x4 q; } o;
#pragma unroll
    for (int e = 0; e < 4; ++e) { o.h[e] = f2bf(f0[e]); o.h[4 + e] = f2bf(f1[e]); }
    ((u32x4*)dst)[i] = o.q;
}

// ---------------------------------------------------------------------------
// Main GEMM (packed bf16 A and B): Out[B,C] = A[B,D] * Bw[C,D]^T
// Double-buffered async LDS B tile; 2 M-fragments per wave so each staged
// B fragment feeds two independent WMMA chains.
// ---------------------------------------------------------------------------
__global__ __launch_bounds__(256)
void gemm_wmma_packed(const unsigned short* __restrict__ Ap,
                      const unsigned short* __restrict__ Bp,
                      float* __restrict__ Out) {
    __shared__ __align__(16) unsigned short ldsB[2][NT * LDSTRIDE];

    const int tid   = threadIdx.x;
    const int wave  = tid >> 5;
    const int lane  = tid & 31;
    const int halfk = lane >> 4;       // 0: lanes hold k 0-7,16-23; 1: k 8-15,24-31
    const int l15   = lane & 15;

    const int mBase = blockIdx.x * MTP; // x fastest -> 16 blocks share one B tile via L2
    const int nBase = blockIdx.y * NT;

    // Staging map: thread covers 5 16B-chunks of the 80 x KC bf16 tile.
    auto stage = [&](int buf, int kc) {
#pragma unroll
        for (int it = 0; it < CHUNKS_PER_THREAD; ++it) {
            const int idx = it * 256 + tid;
            const int n   = idx >> 4;          // KC/8 == 16 chunks per row
            const int j   = idx & 15;
            const unsigned short* g = Bp + (size_t)(nBase + n) * D_K + kc + j * 8;
            unsigned short*       d = &ldsB[buf][n * LDSTRIDE + j * 8];
#if HAVE_ASYNC_LDS
            __builtin_amdgcn_global_load_async_to_lds_b128(
                as_global_v4i(g), as_lds_v4i(d), 0, 0);
#else
            *(u32x4*)d = *(const u32x4*)g;
#endif
        }
    };

    v8f acc0[5], acc1[5];
#pragma unroll
    for (int t = 0; t < 5; ++t)
#pragma unroll
        for (int e = 0; e < 8; ++e) { acc0[t][e] = 0.0f; acc1[t][e] = 0.0f; }

    stage(0, 0);
#if HAVE_ASYNC_LDS
    wait_asynccnt0();
#endif
    __syncthreads();

    // Two per-lane A rows (bf16, L2-resident); fragment = 2 x b128.
    const unsigned short* arow0 =
        Ap + (size_t)(mBase + wave * 32 + l15) * D_K + halfk * 8;
    const unsigned short* arow1 = arow0 + (size_t)16 * D_K;

    for (int i = 0; i < NITER; ++i) {
        const int cur = i & 1;
        if (i + 1 < NITER) stage(cur ^ 1, (i + 1) * KC);   // overlaps with compute below
        const int kc = i * KC;

#pragma unroll
        for (int kk = 0; kk < KC; kk += 32) {
            Frag16 bF[5];
#pragma unroll
            for (int t = 0; t < 5; ++t) {      // issue all ds loads before any WMMA
                const unsigned short* r =
                    &ldsB[cur][(t * 16 + l15) * LDSTRIDE + kk + halfk * 8];
                bF[t].q[0] = *(const u32x4*)r;
                bF[t].q[1] = *(const u32x4*)(r + 16);
            }
            Frag16 aF0, aF1;
            aF0.q[0] = *(const u32x4*)(arow0 + kc + kk);
            aF0.q[1] = *(const u32x4*)(arow0 + kc + kk + 16);
            aF1.q[0] = *(const u32x4*)(arow1 + kc + kk);
            aF1.q[1] = *(const u32x4*)(arow1 + kc + kk + 16);
#pragma unroll
            for (int t = 0; t < 5; ++t) {      // each bF[t] feeds two independent WMMAs
                acc0[t] = __builtin_amdgcn_wmma_f32_16x16x32_bf16(
                    false, aF0.v, false, bF[t].v, (short)0, acc0[t], false, false);
                acc1[t] = __builtin_amdgcn_wmma_f32_16x16x32_bf16(
                    false, aF1.v, false, bF[t].v, (short)0, acc1[t], false, false);
            }
        }
#if HAVE_ASYNC_LDS
        wait_asynccnt0();                       // next buffer landed in LDS
#endif
        __syncthreads();                        // all waves done reading cur + next ready
    }

    // D layout: VGPR g -> row g (lanes 0-15) / g+8 (lanes 16-31); col = lane%16
    const int mRow0 = mBase + wave * 32 + (halfk << 3);
    float* outp0 = Out + (size_t)mRow0 * C_COLS + nBase + l15;
    float* outp1 = outp0 + (size_t)16 * C_COLS;
#pragma unroll
    for (int t = 0; t < 5; ++t)
#pragma unroll
        for (int g = 0; g < 8; ++g) {
            outp0[(size_t)g * C_COLS + t * 16] = acc0[t][g];
            outp1[(size_t)g * C_COLS + t * 16] = acc1[t][g];
        }
}

// ---------------------------------------------------------------------------
// Fallback (workspace too small): read fp32, convert in-registers/at stage.
// ---------------------------------------------------------------------------
__global__ __launch_bounds__(256)
void gemm_wmma_fallback(const float* __restrict__ A,
                        const float* __restrict__ Bf,
                        float* __restrict__ Out) {
    __shared__ __align__(16) unsigned short ldsB[NT * LDSTRIDE];

    const int tid   = threadIdx.x;
    const int wave  = tid >> 5;
    const int lane  = tid & 31;
    const int halfk = lane >> 4;
    const int l15   = lane & 15;
    const int mBase = blockIdx.x * MTF;
    const int nBase = blockIdx.y * NT;

    v8f acc[5];
#pragma unroll
    for (int t = 0; t < 5; ++t)
#pragma unroll
        for (int e = 0; e < 8; ++e) acc[t][e] = 0.0f;

    for (int kc = 0; kc < D_K; kc += KC) {
        __syncthreads();
#pragma unroll
        for (int it = 0; it < CHUNKS_PER_THREAD; ++it) {
            const int idx = it * 256 + tid;
            const int n   = idx >> 4;
            const int j   = idx & 15;
            const f32x4* s = (const f32x4*)(Bf + (size_t)(nBase + n) * D_K + kc + j * 8);
            f32x4 f0 = s[0], f1 = s[1];
            union { unsigned short h[8]; u32x4 q; } o;
#pragma unroll
            for (int e = 0; e < 4; ++e) { o.h[e] = f2bf(f0[e]); o.h[4 + e] = f2bf(f1[e]); }
            *(u32x4*)&ldsB[n * LDSTRIDE + j * 8] = o.q;
        }
        __syncthreads();

        const float* arow = A + (size_t)(mBase + wave * 16 + l15) * D_K + kc + halfk * 8;
#pragma unroll
        for (int kk = 0; kk < KC; kk += 32) {
            Frag16 bF[5];
#pragma unroll
            for (int t = 0; t < 5; ++t) {
                const unsigned short* r = &ldsB[(t * 16 + l15) * LDSTRIDE + kk + halfk * 8];
                bF[t].q[0] = *(const u32x4*)r;
                bF[t].q[1] = *(const u32x4*)(r + 16);
            }
            Frag16 aF;
            {
                const f32x4* p0 = (const f32x4*)(arow + kk);
                const f32x4* p1 = (const f32x4*)(arow + kk + 16);
                f32x4 c00 = p0[0], c01 = p0[1], c10 = p1[0], c11 = p1[1];
#pragma unroll
                for (int e = 0; e < 4; ++e) {
                    aF.h[e]      = f2bf(c00[e]);
                    aF.h[4 + e]  = f2bf(c01[e]);
                    aF.h[8 + e]  = f2bf(c10[e]);
                    aF.h[12 + e] = f2bf(c11[e]);
                }
            }
#pragma unroll
            for (int t = 0; t < 5; ++t)
                acc[t] = __builtin_amdgcn_wmma_f32_16x16x32_bf16(
                    false, aF.v, false, bF[t].v, (short)0, acc[t], false, false);
        }
    }

    const int mRow = mBase + wave * 16 + (halfk << 3);
    float* outp = Out + (size_t)mRow * C_COLS + nBase + l15;
#pragma unroll
    for (int t = 0; t < 5; ++t)
#pragma unroll
        for (int g = 0; g < 8; ++g)
            outp[(size_t)g * C_COLS + t * 16] = acc[t][g];
}

// ---------------------------------------------------------------------------
extern "C" void kernel_launch(void* const* d_in, const int* in_sizes, int n_in,
                              void* d_out, int out_size, void* d_ws, size_t ws_size,
                              hipStream_t stream) {
    (void)in_sizes; (void)n_in; (void)out_size;
    const float* inputs = (const float*)d_in[0];   // [4096, 2048] fp32
    // d_in[1] = targets (int64): unused in forward
    const float* memory = (const float*)d_in[2];   // [50000, 2048] fp32
    float* out = (float*)d_out;                    // [4096, 50000] fp32

    const size_t aElems = (size_t)B_ROWS * D_K;
    const size_t bElems = (size_t)C_COLS * D_K;

    if (ws_size >= (aElems + bElems) * sizeof(unsigned short)) {
        unsigned short* Aw = (unsigned short*)d_ws;
        unsigned short* Bw = Aw + aElems;
        {
            const long long n8 = (long long)aElems / 8;
            pack_bf16_kernel<<<(unsigned)((n8 + 255) / 256), 256, 0, stream>>>(inputs, Aw, n8);
        }
        {
            const long long n8 = (long long)bElems / 8;
            pack_bf16_kernel<<<(unsigned)((n8 + 255) / 256), 256, 0, stream>>>(memory, Bw, n8);
        }
        const dim3 grid(B_ROWS / MTP, C_COLS / NT);      // 16 x 625
        gemm_wmma_packed<<<grid, 256, 0, stream>>>(Aw, Bw, out);
    } else {
        const dim3 grid(B_ROWS / MTF, C_COLS / NT);      // 32 x 625
        gemm_wmma_fallback<<<grid, 256, 0, stream>>>(inputs, memory, out);
    }
}
